// GAT_Model_77756087927555
// MI455X (gfx1250) — compile-verified, matrix operand order checked by reading, not run on hardware
//
#include <hip/hip_runtime.h>
#include <hip/hip_bf16.h>

#define N_NODES 20000
#define E_RAW   320000
#define ET      (E_RAW + N_NODES)   // with self loops = 340000
#define NGRAPH  64
#define D_IN    128
#define HID     256
#define HEADS   8
#define CH      32                  // per-head dim
#define NLAYERS 4
#define NEG_SLOPE 0.2f
#define BN_EPS  1e-5f

typedef __attribute__((ext_vector_type(16))) __bf16 v16bf;
typedef __attribute__((ext_vector_type(8)))  __bf16 v8bf;
typedef __attribute__((ext_vector_type(8)))  float  v8f;

// ---------------------------------------------------------------------------
// WMMA GEMM:  Out[M,NO] = act( A[M,K] @ W[K,NO] + bias )
//   A  : bf16 row-major [M,K]
//   Wt : bf16 TRANSPOSED weights [NO,K]
// Each lane's fragment (ISA 7.12.2, 16-bit A 16x32) is two contiguous 8-bf16
// runs: K = {8h..8h+7} and {16+8h..23+8h}  ->  2x global_load_b128 per frag.
// Block = 64 threads (2 waves); each wave owns a 16x64 output tile.
// K templated so the loop fully unrolls (K/32 steps x 4 WMMA each).
// ---------------------------------------------------------------------------
template<int K>
__global__ void wmma_gemm_bf16(const __bf16* __restrict__ A,
                               const __bf16* __restrict__ Wt,
                               const float* __restrict__ bias,
                               float* __restrict__ Out,
                               __bf16* __restrict__ OutBf,
                               int ldo, int act)
{
    const int lane = threadIdx.x & 31;
    const int wv   = threadIdx.x >> 5;
    const int half = lane >> 4;          // K-half selector
    const int idx  = lane & 15;          // row (A) / col (B,D) within tile
    const int row0 = (blockIdx.x * (blockDim.x >> 5) + wv) * 16;
    const int col0 = blockIdx.y * 64;

    v8f acc[4] = {{}, {}, {}, {}};

    const __bf16* ap = A + (size_t)(row0 + idx) * K + 8 * half;

#pragma unroll
    for (int k0 = 0; k0 < K; k0 += 32) {
        union { v16bf v; v8bf h[2]; } af;
        af.h[0] = *(const v8bf*)(ap + k0);
        af.h[1] = *(const v8bf*)(ap + k0 + 16);
#pragma unroll
        for (int t = 0; t < 4; ++t) {
            const __bf16* bp =
                Wt + (size_t)(col0 + t * 16 + idx) * K + k0 + 8 * half;
            union { v16bf v; v8bf h[2]; } bw;
            bw.h[0] = *(const v8bf*)(bp);
            bw.h[1] = *(const v8bf*)(bp + 16);
            acc[t] = __builtin_amdgcn_wmma_f32_16x16x32_bf16(
                         false, af.v, false, bw.v, (short)0, acc[t], false, false);
        }
    }

    // ---- store D: VGPR r holds (row = r + 8*half, col = idx) ----
#pragma unroll
    for (int t = 0; t < 4; ++t) {
        const int col = col0 + t * 16 + idx;
        const float bv = bias ? bias[col] : 0.0f;
#pragma unroll
        for (int r = 0; r < 8; ++r) {
            const int row = row0 + r + 8 * half;
            float v = acc[t][r] + bv;
            if (act) v = fmaxf(v, 0.0f);
            if (Out)   Out[(size_t)row * ldo + col] = v;
            if (OutBf) OutBf[(size_t)row * ldo + col] = (__bf16)v;
        }
    }
}

// f32 -> bf16 elementwise
__global__ void k_f32_to_bf16(const float* __restrict__ in, __bf16* __restrict__ out,
                              int n)
{
    int t = blockIdx.x * blockDim.x + threadIdx.x;
    if (t < n) out[t] = (__bf16)in[t];
}

// W[K,NO] f32  ->  Wt[NO,K] bf16 (transpose + convert; one-shot, tiny)
__global__ void k_transpose_w(const float* __restrict__ W, __bf16* __restrict__ Wt,
                              int K, int NO)
{
    int t = blockIdx.x * blockDim.x + threadIdx.x;
    if (t >= K * NO) return;
    const int n = t / K, k = t - n * K;
    Wt[t] = (__bf16)W[(size_t)k * NO + n];
}

// ---------------------------------------------------------------------------
// Edge helpers: edge_index is [2,E]; self loops appended virtually.
// ---------------------------------------------------------------------------
__device__ __forceinline__ int eSrc(const int* ei, int e) {
    return (e < E_RAW) ? ei[e] : (e - E_RAW);
}
__device__ __forceinline__ int eDst(const int* ei, int e) {
    return (e < E_RAW) ? ei[E_RAW + e] : (e - E_RAW);
}

__device__ __forceinline__ void atomicMaxF(float* addr, float val) {
    unsigned int* ua = (unsigned int*)addr;
    unsigned int cur = *ua;
    while (__uint_as_float(cur) < val) {
        unsigned int assumed = cur;
        cur = atomicCAS(ua, assumed, __float_as_uint(val));
        if (cur == assumed) break;
    }
}

// alpha_s/alpha_d[n,h] = sum_c hproj[n,h*32+c] * a_{src,dst}[h*32+c]
__global__ void k_alpha(const float* __restrict__ hp,
                        const float* __restrict__ a_src,
                        const float* __restrict__ a_dst,
                        float* __restrict__ as, float* __restrict__ ad)
{
    int t = blockIdx.x * blockDim.x + threadIdx.x;   // n*HEADS + h
    if (t >= N_NODES * HEADS) return;
    const int n = t >> 3, h = t & 7;
    const float* v = hp + (size_t)n * HID + h * CH;
    float s1 = 0.f, s2 = 0.f;
#pragma unroll
    for (int c = 0; c < CH; ++c) {
        float x = v[c];
        s1 += x * a_src[h * CH + c];
        s2 += x * a_dst[h * CH + c];
    }
    as[t] = s1; ad[t] = s2;
}

__global__ void k_init_mz(float* __restrict__ m, float* __restrict__ z)
{
    int t = blockIdx.x * blockDim.x + threadIdx.x;
    if (t >= N_NODES * HEADS) return;
    m[t] = -3.402823e38f;
    z[t] = 0.f;
}

// leaky-relu logit per edge/head + segment max into m[dst]
__global__ void k_logit_max(const float* __restrict__ as, const float* __restrict__ ad,
                            const int* __restrict__ ei,
                            float* __restrict__ logits, float* __restrict__ m)
{
    int e = blockIdx.x * blockDim.x + threadIdx.x;
    if (e >= ET) return;
    const int s = eSrc(ei, e), d = eDst(ei, e);
#pragma unroll
    for (int h = 0; h < HEADS; ++h) {
        float a = as[s * HEADS + h] + ad[d * HEADS + h];
        a = (a > 0.f) ? a : NEG_SLOPE * a;
        logits[(size_t)e * HEADS + h] = a;
        atomicMaxF(&m[d * HEADS + h], a);
    }
}

// e = exp(logit - m[dst]); segment sum into z[dst]
__global__ void k_expsum(const int* __restrict__ ei, const float* __restrict__ m,
                         float* __restrict__ logits, float* __restrict__ z)
{
    int e = blockIdx.x * blockDim.x + threadIdx.x;
    if (e >= ET) return;
    const int d = eDst(ei, e);
#pragma unroll
    for (int h = 0; h < HEADS; ++h) {
        float ee = __expf(logits[(size_t)e * HEADS + h] - m[d * HEADS + h]);
        logits[(size_t)e * HEADS + h] = ee;
        atomicAdd(&z[d * HEADS + h], ee);
    }
}

// init aggregation buffer with conv bias (folds "+ b")
__global__ void k_init_agg(float* __restrict__ agg, const float* __restrict__ b)
{
    size_t t = (size_t)blockIdx.x * blockDim.x + threadIdx.x;
    if (t >= (size_t)N_NODES * HID) return;
    agg[t] = b[t & (HID - 1)];
}

// agg[dst, c] += hproj[src, c] * (e / z[dst, head(c)])  — one block per edge
__global__ void k_aggregate(const float* __restrict__ hp, const float* __restrict__ logits,
                            const float* __restrict__ z, const int* __restrict__ ei,
                            float* __restrict__ agg)
{
    const int e = blockIdx.x;
    const int c = threadIdx.x;          // 0..255
    const int h = c >> 5;
    const int s = eSrc(ei, e), d = eDst(ei, e);
    const float w = logits[(size_t)e * HEADS + h] / z[d * HEADS + h];
    atomicAdd(&agg[(size_t)d * HID + c], hp[(size_t)s * HID + c] * w);
}

// per-channel mean / inv-std over N_NODES rows (one block per channel)
__global__ void k_bn_stats(const float* __restrict__ x,
                           float* __restrict__ mean, float* __restrict__ rstd)
{
    __shared__ float ssum[256];
    __shared__ float ssq[256];
    const int c = blockIdx.x;
    float s = 0.f, q = 0.f;
    for (int n = threadIdx.x; n < N_NODES; n += 256) {
        float v = x[(size_t)n * HID + c];
        s += v; q += v * v;
    }
    ssum[threadIdx.x] = s; ssq[threadIdx.x] = q;
    __syncthreads();
    for (int off = 128; off > 0; off >>= 1) {
        if (threadIdx.x < off) {
            ssum[threadIdx.x] += ssum[threadIdx.x + off];
            ssq[threadIdx.x]  += ssq[threadIdx.x + off];
        }
        __syncthreads();
    }
    if (threadIdx.x == 0) {
        float mu = ssum[0] / (float)N_NODES;
        float var = ssq[0] / (float)N_NODES - mu * mu;
        mean[c] = mu;
        rstd[c] = rsqrtf(var + BN_EPS);
    }
}

// BN + ReLU; writes both f32 (for pooling) and bf16 (next layer's GEMM input)
__global__ void k_bn_apply_relu(const float* __restrict__ x,
                                const float* __restrict__ g, const float* __restrict__ b,
                                const float* __restrict__ mean, const float* __restrict__ rstd,
                                float* __restrict__ out, __bf16* __restrict__ outbf)
{
    size_t t = (size_t)blockIdx.x * blockDim.x + threadIdx.x;
    if (t >= (size_t)N_NODES * HID) return;
    const int c = (int)(t & (HID - 1));
    float v = g[c] * (x[t] - mean[c]) * rstd[c] + b[c];
    v = fmaxf(v, 0.f);
    out[t] = v;
    outbf[t] = (__bf16)v;
}

__global__ void k_init_pool(float* __restrict__ sums, float* __restrict__ cnt)
{
    int t = blockIdx.x * blockDim.x + threadIdx.x;
    if (t < NGRAPH * HID) sums[t] = 0.f;
    if (t < NGRAPH) cnt[t] = 0.f;
}

__global__ void k_pool(const float* __restrict__ h, const int* __restrict__ batch,
                       float* __restrict__ sums, float* __restrict__ cnt)
{
    const int n = blockIdx.x;
    const int c = threadIdx.x;
    const int g = batch[n];
    atomicAdd(&sums[(size_t)g * HID + c], h[(size_t)n * HID + c]);
    if (c == 0) atomicAdd(&cnt[g], 1.0f);
}

__global__ void k_pool_mean(const float* __restrict__ sums, const float* __restrict__ cnt,
                            __bf16* __restrict__ pooledbf)
{
    int t = blockIdx.x * blockDim.x + threadIdx.x;
    if (t >= NGRAPH * HID) return;
    pooledbf[t] = (__bf16)(sums[t] / fmaxf(cnt[t >> 8], 1.0f));
}

__global__ void k_final(const float* __restrict__ hidden, const float* __restrict__ W2,
                        const float* __restrict__ b2, float* __restrict__ out)
{
    int g = threadIdx.x;
    if (g >= NGRAPH) return;
    float s = b2[0];
#pragma unroll 8
    for (int j = 0; j < HID / 2; ++j) s += hidden[g * (HID / 2) + j] * W2[j];
    out[g] = s;
}

// ---------------------------------------------------------------------------
extern "C" void kernel_launch(void* const* d_in, const int* in_sizes, int n_in,
                              void* d_out, int out_size, void* d_ws, size_t ws_size,
                              hipStream_t stream)
{
    (void)in_sizes; (void)n_in; (void)out_size; (void)ws_size;

    const float* x        = (const float*)d_in[0];
    const int*   ei       = (const int*)  d_in[1];
    const int*   batch    = (const int*)  d_in[2];
    const float* proj_W   = (const float*)d_in[3];
    const float* proj_b   = (const float*)d_in[4];
    const float* lin_W    = (const float*)d_in[5];   // [L,256,256]
    const float* att_src  = (const float*)d_in[6];   // [L,8,32]
    const float* att_dst  = (const float*)d_in[7];
    const float* conv_b   = (const float*)d_in[8];   // [L,256]
    const float* bn_g     = (const float*)d_in[9];
    const float* bn_b     = (const float*)d_in[10];
    const float* pred_W1  = (const float*)d_in[11];  // [256,128]
    const float* pred_b1  = (const float*)d_in[12];
    const float* pred_W2  = (const float*)d_in[13];  // [128,1]
    const float* pred_b2  = (const float*)d_in[14];
    float* out = (float*)d_out;

    // ---- workspace carve: f32 region then 64B-aligned bf16 region ----
    float* p = (float*)d_ws;
    float* hA     = p; p += (size_t)N_NODES * HID;        // current features f32
    float* hP     = p; p += (size_t)N_NODES * HID;        // projected features f32
    float* agg    = p; p += (size_t)N_NODES * HID;        // aggregation accumulator
    float* as     = p; p += (size_t)N_NODES * HEADS;
    float* ad     = p; p += (size_t)N_NODES * HEADS;
    float* m      = p; p += (size_t)N_NODES * HEADS;
    float* z      = p; p += (size_t)N_NODES * HEADS;
    float* logits = p; p += (size_t)ET * HEADS;
    float* mean   = p; p += HID;
    float* rstd   = p; p += HID;
    float* sums   = p; p += NGRAPH * HID;
    float* cnt    = p; p += NGRAPH;
    float* hidden = p; p += NGRAPH * (HID / 2);

    uintptr_t bb = ((uintptr_t)p + 63) & ~(uintptr_t)63;
    __bf16* q = (__bf16*)bb;
    __bf16* xbf      = q; q += (size_t)N_NODES * D_IN;
    __bf16* hAbf     = q; q += (size_t)N_NODES * HID;
    __bf16* pooledbf = q; q += NGRAPH * HID;
    __bf16* WtProj   = q; q += (size_t)HID * D_IN;              // [256,128]
    __bf16* WtLin    = q; q += (size_t)NLAYERS * HID * HID;     // [4][256,256]
    __bf16* WtP1     = q; q += (size_t)(HID / 2) * HID;         // [128,256]

    const int nhBlocks = (N_NODES * HEADS + 255) / 256;
    const int edBlocks = (ET + 255) / 256;
    const int elBlocks = (int)(((size_t)N_NODES * HID + 255) / 256);

    // ---- one-shot conversions: activations to bf16, weights to bf16^T ----
    k_f32_to_bf16<<<(N_NODES * D_IN + 255) / 256, 256, 0, stream>>>(x, xbf, N_NODES * D_IN);
    k_transpose_w<<<(D_IN * HID + 255) / 256, 256, 0, stream>>>(proj_W, WtProj, D_IN, HID);
    for (int l = 0; l < NLAYERS; ++l)
        k_transpose_w<<<(HID * HID + 255) / 256, 256, 0, stream>>>(
            lin_W + (size_t)l * HID * HID, WtLin + (size_t)l * HID * HID, HID, HID);
    k_transpose_w<<<(HID * (HID / 2) + 255) / 256, 256, 0, stream>>>(
        pred_W1, WtP1, HID, HID / 2);

    // input projection: hA(+bf16) = relu(x @ proj_W + proj_b)  [20000,128]x[128,256]
    wmma_gemm_bf16<D_IN><<<dim3(N_NODES / 32, HID / 64), 64, 0, stream>>>(
        xbf, WtProj, proj_b, hA, hAbf, HID, 1);

    for (int l = 0; l < NLAYERS; ++l) {
        const __bf16* Wt = WtLin + (size_t)l * HID * HID;
        const float* aS = att_src + (size_t)l * HEADS * CH;
        const float* aD = att_dst + (size_t)l * HEADS * CH;
        const float* cb = conv_b + (size_t)l * HID;
        const float* g  = bn_g   + (size_t)l * HID;
        const float* b  = bn_b   + (size_t)l * HID;

        // hP = hA @ W   [20000,256]x[256,256]
        wmma_gemm_bf16<HID><<<dim3(N_NODES / 32, HID / 64), 64, 0, stream>>>(
            hAbf, Wt, nullptr, hP, nullptr, HID, 0);

        k_alpha<<<nhBlocks, 256, 0, stream>>>(hP, aS, aD, as, ad);
        k_init_mz<<<nhBlocks, 256, 0, stream>>>(m, z);
        k_logit_max<<<edBlocks, 256, 0, stream>>>(as, ad, ei, logits, m);
        k_expsum<<<edBlocks, 256, 0, stream>>>(ei, m, logits, z);
        k_init_agg<<<elBlocks, 256, 0, stream>>>(agg, cb);
        k_aggregate<<<ET, 256, 0, stream>>>(hP, logits, z, ei, agg);
        k_bn_stats<<<HID, 256, 0, stream>>>(agg, mean, rstd);
        k_bn_apply_relu<<<elBlocks, 256, 0, stream>>>(agg, g, b, mean, rstd, hA, hAbf);
    }

    // global mean pool
    k_init_pool<<<(NGRAPH * HID + 255) / 256, 256, 0, stream>>>(sums, cnt);
    k_pool<<<N_NODES, HID, 0, stream>>>(hA, batch, sums, cnt);
    k_pool_mean<<<(NGRAPH * HID + 255) / 256, 256, 0, stream>>>(sums, cnt, pooledbf);

    // predictor: hidden = relu(pooled @ W1 + b1)  [64,256]x[256,128]
    wmma_gemm_bf16<HID><<<dim3(NGRAPH / 32, (HID / 2) / 64), 64, 0, stream>>>(
        pooledbf, WtP1, pred_b1, hidden, nullptr, HID / 2, 1);
    // out = hidden @ W2 + b2
    k_final<<<1, 64, 0, stream>>>(hidden, pred_W2, pred_b2, out);
}
